// QkvRelativeAttention_35270271434814
// MI455X (gfx1250) — matrix operand order repaired and checked by reading, not compiled, not run on hardware
//
#include <hip/hip_runtime.h>

// Problem constants from the reference (fixed by setup_inputs):
// B=2, Q=2048, K=2048, H=16, D=64, R=32
#define B_  2
#define Q_  2048
#define K_  2048
#define H_  16
#define D_  64
#define R_  32

typedef __attribute__((ext_vector_type(16))) __bf16        v16bf;
typedef __attribute__((ext_vector_type(8)))  float         v8f;
typedef __attribute__((ext_vector_type(8)))  unsigned int  v8u;

#define LOG2E     1.4426950408889634f
#define SCALE_L2  (0.125f * LOG2E)        /* (1/sqrt(64)) * log2(e) */
#define MASKADD_L2 (-10000.0f * LOG2E)    /* mask adder in log2 domain */

static __device__ __forceinline__ unsigned int pack2bf(float lo, float hi) {
    unsigned short a = __builtin_bit_cast(unsigned short, (__bf16)lo);
    unsigned short b = __builtin_bit_cast(unsigned short, (__bf16)hi);
    return (unsigned int)a | ((unsigned int)b << 16);
}

// Flash-attention with relative-position gather.
// 4 waves / 128-thread block; each wave owns a 16-row q tile; all waves share
// (b,h) and iterate k tiles in lockstep, so K/V tiles are staged ONCE per
// block as packed bf16 in LDS (4x less global traffic + f32->bf16 VALU than
// per-wave staging).
//
// LDS:
//   kT  [32 k][32 d2] u32  = (bf16 K[k][2*d2], bf16 K[k][2*d2+1])        4KB
//   vP  [16 k2][64 d] u32  = (bf16 V[2*k2][d], bf16 V[2*k2+1][d])        4KB
//   per wave: relL f32[16 q][32 r] (2KB) + pL f32[16 q][32 k] (2KB)     16KB
__global__ __launch_bounds__(128)
void qkv_rel_attn_kernel(const float* __restrict__ qptr,
                         const float* __restrict__ kptr,
                         const float* __restrict__ vptr,
                         const int*   __restrict__ maskptr,
                         const int*   __restrict__ idsptr,
                         const float* __restrict__ embptr,
                         const float* __restrict__ biasptr,
                         float*       __restrict__ outptr) {
    __shared__ unsigned int smemU[2048];     // kT[0..1023], vP[1024..2047]
    __shared__ float        smemF[4 * 1024]; // per-wave relL(512) + pL(512)

    const int tid  = threadIdx.x;
    const int wave = tid >> 5;
    const int lane = tid & 31;
    const int col  = lane & 15;
    const int half = lane >> 4;

    const int qtiles = Q_ / 64;
    int bi = blockIdx.x;
    const int qt = bi % qtiles; bi /= qtiles;
    const int h  = bi % H_;     bi /= H_;
    const int b  = bi;
    const int qb = qt * 64 + wave * 16;

    unsigned int* kT   = smemU;
    unsigned int* vP   = smemU + 1024;
    float*        relL = smemF + wave * 1024;
    float*        pL   = relL + 512;

    // ---------------- Q A-fragments (16x32 bf16), d-chunks [0,32) and [32,64)
    v16bf aq[2];
    {
        const float* qrow = qptr + (((size_t)b * Q_ + qb + col) * H_ + h) * D_;
#pragma unroll
        for (int f = 0; f < 2; ++f) {
            const int dc = f * 32;
#pragma unroll
            for (int e = 0; e < 16; ++e) {
                const int kl = half * 16 + e;
                aq[f][e] = (__bf16)qrow[dc + kl];
            }
        }
    }

    // ---------------- all_rel[16 q][32 r] = Q . E^T + bias  -> LDS (per wave)
#pragma unroll
    for (int rb = 0; rb < 2; ++rb) {
        v8f acc = {};
#pragma unroll
        for (int f = 0; f < 2; ++f) {
            const int dc = f * 32;
            const float* erow = embptr + (((size_t)(rb * 16 + col)) * H_ + h) * D_ + dc + half * 16;
            v16bf bfrag;
#pragma unroll
            for (int e = 0; e < 16; ++e) bfrag[e] = (__bf16)erow[e];
            acc = __builtin_amdgcn_wmma_f32_16x16x32_bf16(
                false, aq[f], false, bfrag, (short)0, acc, false, false);
        }
        const float bv = biasptr[(rb * 16 + col) * H_ + h];
#pragma unroll
        for (int j = 0; j < 8; ++j)
            relL[(half * 8 + j) * 32 + rb * 16 + col] = acc[j] + bv;
    }
    __builtin_amdgcn_wave_barrier();

    // ---------------- flash-attention state
    v8f o[4] = {{}, {}, {}, {}};
    float m[8], l[8];
#pragma unroll
    for (int j = 0; j < 8; ++j) { m[j] = -1e30f; l[j] = 0.0f; }

    const int* idrow0 = idsptr  + ((size_t)b * Q_ + qb) * K_;
    const int* mkrow0 = maskptr + ((size_t)b * Q_ + qb) * K_;

    // staging assignments (block-cooperative)
    const int kr = tid >> 2;            // K stage: row 0..31
    const int kc = (tid & 3) * 16;      // K stage: d chunk base
    const int vk2 = tid >> 3;           // V stage: k-pair 0..15
    const int vdb = (tid & 7) * 8;      // V stage: d chunk base

    for (int kb = 0; kb < K_; kb += 32) {
        // ---- stage K tile (32 x 64 f32 -> packed bf16 d-pairs), once per block
        {
            const float* src = kptr + (((size_t)b * K_ + kb + kr) * H_ + h) * D_ + kc;
            if (kb + 32 < K_) __builtin_prefetch(src + 32 * H_ * D_, 0, 0);
            const float4 f0 = *(const float4*)(src + 0);
            const float4 f1 = *(const float4*)(src + 4);
            const float4 f2 = *(const float4*)(src + 8);
            const float4 f3 = *(const float4*)(src + 12);
            uint4 p0, p1;
            p0.x = pack2bf(f0.x, f0.y); p0.y = pack2bf(f0.z, f0.w);
            p0.z = pack2bf(f1.x, f1.y); p0.w = pack2bf(f1.z, f1.w);
            p1.x = pack2bf(f2.x, f2.y); p1.y = pack2bf(f2.z, f2.w);
            p1.z = pack2bf(f3.x, f3.y); p1.w = pack2bf(f3.z, f3.w);
            unsigned int* dst = kT + kr * 32 + (kc >> 1);
            *(uint4*)(dst + 0) = p0;
            *(uint4*)(dst + 4) = p1;
        }
        // ---- stage V tile (32 x 64 f32 -> packed bf16 k-pairs), once per block
        {
            const float* s0 = vptr + (((size_t)b * K_ + kb + 2 * vk2) * H_ + h) * D_ + vdb;
            const float* s1 = s0 + (size_t)H_ * D_;
            if (kb + 32 < K_) __builtin_prefetch(s0 + 32 * H_ * D_, 0, 0);
            const float4 f0 = *(const float4*)(s0 + 0);
            const float4 f1 = *(const float4*)(s0 + 4);
            const float4 g0 = *(const float4*)(s1 + 0);
            const float4 g1 = *(const float4*)(s1 + 4);
            uint4 p0, p1;
            p0.x = pack2bf(f0.x, g0.x); p0.y = pack2bf(f0.y, g0.y);
            p0.z = pack2bf(f0.z, g0.z); p0.w = pack2bf(f0.w, g0.w);
            p1.x = pack2bf(f1.x, g1.x); p1.y = pack2bf(f1.y, g1.y);
            p1.z = pack2bf(f1.z, g1.z); p1.w = pack2bf(f1.w, g1.w);
            unsigned int* dst = vP + vk2 * 64 + vdb;
            *(uint4*)(dst + 0) = p0;
            *(uint4*)(dst + 4) = p1;
        }
        __syncthreads();

        // ---- S = Q . K^T : two 16x16 accumulators (keys kb+0..15, kb+16..31)
        v8f s[2] = {{}, {}};
#pragma unroll
        for (int kcb = 0; kcb < 2; ++kcb) {
#pragma unroll
            for (int f = 0; f < 2; ++f) {
                const unsigned int* kp = kT + (kcb * 16 + col) * 32 + ((f * 32 + half * 16) >> 1);
                v8u bu;
#pragma unroll
                for (int j = 0; j < 8; ++j) bu[j] = kp[j];
                const v16bf bfrag = __builtin_bit_cast(v16bf, bu);
                s[kcb] = __builtin_amdgcn_wmma_f32_16x16x32_bf16(
                    false, aq[f], false, bfrag, (short)0, s[kcb], false, false);
            }
        }

        // ---- gather rel + mask, scale (log2 domain), online softmax; P -> LDS
#pragma unroll
        for (int j = 0; j < 8; ++j) {
            const int row  = half * 8 + j;
            const size_t qoff = (size_t)row * K_;
            const int k0 = kb + col, k1 = kb + 16 + col;
            const int id0 = idrow0[qoff + k0];
            const int id1 = idrow0[qoff + k1];
            const int mk0 = mkrow0[qoff + k0];
            const int mk1 = mkrow0[qoff + k1];
            const float r0 = relL[row * 32 + id0];
            const float r1 = relL[row * 32 + id1];
            float s0 = (s[0][j] + r0) * SCALE_L2 + (mk0 ? 0.0f : MASKADD_L2);
            float s1 = (s[1][j] + r1) * SCALE_L2 + (mk1 ? 0.0f : MASKADD_L2);
            float mx = fmaxf(s0, s1);
            mx = fmaxf(mx, __shfl_xor(mx, 1, 32));
            mx = fmaxf(mx, __shfl_xor(mx, 2, 32));
            mx = fmaxf(mx, __shfl_xor(mx, 4, 32));
            mx = fmaxf(mx, __shfl_xor(mx, 8, 32));
            const float mn    = fmaxf(m[j], mx);
            const float alpha = exp2f(m[j] - mn);
            const float p0    = exp2f(s0 - mn);
            const float p1    = exp2f(s1 - mn);
            l[j] = l[j] * alpha + p0 + p1;
            m[j] = mn;
#pragma unroll
            for (int blk = 0; blk < 4; ++blk) o[blk][j] *= alpha;
            pL[row * 32 + col]      = p0;
            pL[row * 32 + 16 + col] = p1;
        }
        __builtin_amdgcn_wave_barrier();

        // ---- P as A-fragment (16x32 bf16) via per-wave LDS transpose
        v16bf ap;
#pragma unroll
        for (int e = 0; e < 16; ++e) {
            const int kl = half * 16 + e;
            ap[e] = (__bf16)pL[col * 32 + kl];
        }

        // ---- O += P . V  (4 d-blocks of 16)
#pragma unroll
        for (int blk = 0; blk < 4; ++blk) {
            v8u bu;
#pragma unroll
            for (int j = 0; j < 8; ++j)
                bu[j] = vP[(half * 8 + j) * 64 + blk * 16 + col];
            const v16bf bvf = __builtin_bit_cast(v16bf, bu);
            o[blk] = __builtin_amdgcn_wmma_f32_16x16x32_bf16(
                false, ap, false, bvf, (short)0, o[blk], false, false);
        }
        __syncthreads();   // all waves done reading kT/vP before next staging
    }

    // ---------------- finalize: divide by row sums and store
#pragma unroll
    for (int j = 0; j < 8; ++j) {
        float ls = l[j];
        ls += __shfl_xor(ls, 1, 32);
        ls += __shfl_xor(ls, 2, 32);
        ls += __shfl_xor(ls, 4, 32);
        ls += __shfl_xor(ls, 8, 32);
        const float inv = 1.0f / ls;
        float* orow = outptr + (((size_t)b * Q_ + qb + half * 8 + j) * H_ + h) * D_;
#pragma unroll
        for (int blk = 0; blk < 4; ++blk)
            orow[blk * 16 + col] = o[blk][j] * inv;
    }
}

extern "C" void kernel_launch(void* const* d_in, const int* in_sizes, int n_in,
                              void* d_out, int out_size, void* d_ws, size_t ws_size,
                              hipStream_t stream) {
    (void)in_sizes; (void)n_in; (void)d_ws; (void)ws_size; (void)out_size;
    const float* q    = (const float*)d_in[0];
    const float* k    = (const float*)d_in[1];
    const float* v    = (const float*)d_in[2];
    const int*   mask = (const int*)d_in[3];
    const int*   ids  = (const int*)d_in[4];
    const float* emb  = (const float*)d_in[5];
    const float* bias = (const float*)d_in[6];
    float*       out  = (float*)d_out;

    const int blocks = B_ * H_ * (Q_ / 64);   // 1024
    qkv_rel_attn_kernel<<<blocks, 128, 0, stream>>>(q, k, v, mask, ids, emb, bias, out);
}
